// LSTM_29042568856007
// MI455X (gfx1250) — matrix operand order repaired and checked by reading, not compiled, not run on hardware
//
#include <hip/hip_runtime.h>
#include <hip/hip_bf16.h>
#include <math.h>

// ---- problem constants (match reference) ----
#define Bsz   64
#define Tlen  4096
#define Din   256
#define Hdim  256
#define K2    512            // Din + Hdim (fused GEMM contraction)
#define NWG   16             // persistent workgroups
#define HW    16             // hidden units per WG (Hdim / NWG)
#define GR    64             // gate rows per WG = 4 * HW
#define AS    520            // A row stride in halves (512 + 8 pad)
#define WSd   520            // W row stride in halves
#define GS    68             // gate buffer row stride in floats

typedef __attribute__((ext_vector_type(16))) _Float16 v16h;
typedef __attribute__((ext_vector_type(8)))  _Float16 v8h;
typedef __attribute__((ext_vector_type(4)))  _Float16 v4h;
typedef __attribute__((ext_vector_type(8)))  float    v8f;

__device__ __forceinline__ float sigmoidf_fast(float x) {
    return 1.0f / (1.0f + __expf(-x));
}
__device__ __forceinline__ float tanhf_fast(float x) {
    // tanh(x) = 1 - 2/(exp(2x)+1); saturates correctly at +/-1
    return 1.0f - 2.0f / (__expf(2.0f * x) + 1.0f);
}

// Zero the barrier counter and the h ping-pong buffers each launch
// (ws is poisoned once by the harness; graph replays re-run this kernel).
__global__ void lstm_init(unsigned* __restrict__ bar, _Float16* __restrict__ hbuf) {
    if (blockIdx.x == 0 && threadIdx.x == 0) *bar = 0u;
    int i = blockIdx.x * blockDim.x + threadIdx.x;
    const int n = 2 * Bsz * Hdim;
    for (; i < n; i += gridDim.x * blockDim.x) hbuf[i] = (_Float16)0.0f;
}

__global__ __launch_bounds__(256, 1)
void lstm_persistent(const float* __restrict__ xs,
                     const float* __restrict__ Wih,
                     const float* __restrict__ Whh,
                     const float* __restrict__ bias,
                     float* __restrict__ out,
                     _Float16* __restrict__ hbuf,
                     unsigned* __restrict__ bar)
{
    __shared__ __align__(16) _Float16 As[2][Bsz][AS]; // double-buffered x_t || h_t
    __shared__ __align__(16) _Float16 Ws[GR][WSd];    // weight slice W[n][k], f16
    __shared__ float Gl[Bsz][GS];                     // fp32 gate outputs [64][64+pad]
    __shared__ float Cl[Bsz][HW];                     // cell state slice
    __shared__ float Bl[GR];                          // bias slice

    const int wg   = blockIdx.x;        // 0..15
    const int tid  = threadIdx.x;       // 0..255
    const int lane = tid & 31;
    const int wave = tid >> 5;          // 0..7
    const int l16  = lane & 15;
    const int lhi  = lane >> 4;         // 0 or 1

    // ---- one-time: load + convert weight slice into LDS (f16) ----
    // local gate row n = gate*HW + j  ->  global row = gate*Hdim + wg*HW + j
    for (int i = tid; i < GR * 64; i += 256) {      // 64 rows x 64 blocks of 8 floats
        const int n  = i >> 6;
        const int kb = (i & 63) * 8;
        const int gate = n >> 4;
        const int j    = n & 15;
        const int grow = gate * Hdim + wg * HW + j;
        const float* src = (kb < Din) ? &Wih[grow * Din + kb]
                                      : &Whh[grow * Hdim + (kb - Din)];
        const float4 f0 = *(const float4*)(src);
        const float4 f1 = *(const float4*)(src + 4);
        v8h hv;
        hv[0] = (_Float16)f0.x; hv[1] = (_Float16)f0.y;
        hv[2] = (_Float16)f0.z; hv[3] = (_Float16)f0.w;
        hv[4] = (_Float16)f1.x; hv[5] = (_Float16)f1.y;
        hv[6] = (_Float16)f1.z; hv[7] = (_Float16)f1.w;
        *(v8h*)&Ws[n][kb] = hv;
    }
    if (tid < GR) {
        const int gate = tid >> 4, j = tid & 15;
        Bl[tid] = bias[gate * Hdim + wg * HW + j];
    }
    for (int i = tid; i < Bsz * HW; i += 256) Cl[i / HW][i % HW] = 0.0f;

    // ---- prologue: stage x_0 into As[0] x-region ----
    #pragma unroll
    for (int u = 0; u < 16; ++u) {
        const int i  = tid + (u << 8);
        const int bb = i >> 6;
        const int kb = (i & 63) << 2;
        const float4 f = *(const float4*)&xs[((size_t)bb * Tlen + 0) * Din + kb];
        v4h hv;
        hv[0] = (_Float16)f.x; hv[1] = (_Float16)f.y;
        hv[2] = (_Float16)f.z; hv[3] = (_Float16)f.w;
        *(v4h*)&As[0][bb][kb] = hv;
    }
    __syncthreads();

    // wave tile assignment: output tile grid = 4 batch-tiles x 4 gate-tiles
    const int gn  = wave & 3;            // gate tile (== gate index, 16 rows)
    const int bm0 = (wave >> 2) << 1;    // this wave: batch tiles bm0, bm0+1

    for (int t = 0; t < Tlen; ++t) {
        const int q = t & 1;
        __threadfence();                 // acquire prior step's h (agent-scope inv)

        // ---- issue global loads for x_{t+1} (consumed after the K-loop) ----
        float4 xr[16];
        if (t + 1 < Tlen) {
            #pragma unroll
            for (int u = 0; u < 16; ++u) {
                const int i  = tid + (u << 8);
                const int bb = i >> 6;
                const int kb = (i & 63) << 2;
                xr[u] = *(const float4*)&xs[((size_t)bb * Tlen + (t + 1)) * Din + kb];
            }
        }

        // ---- stage h_t into As[q] h-region (already f16, L2-resident) ----
        const _Float16* hsrc = hbuf + q * (Bsz * Hdim);
        for (int i = tid; i < (Bsz * Hdim) / 8; i += 256) {
            const int bb = i >> 5;
            const int kb = (i & 31) << 3;
            *(v8h*)&As[q][bb][Din + kb] = *(const v8h*)&hsrc[bb * Hdim + kb];
        }
        __syncthreads();

        // ---- fused GEMM: gates[64 x 64] = A[64 x 512] * W^T slice + bias ----
        const float bv = Bl[gn * 16 + l16];   // C/D layout: N == l16 for all 8 regs
        v8f acc0, acc1;
        #pragma unroll
        for (int r = 0; r < 8; ++r) { acc0[r] = bv; acc1[r] = bv; }

        for (int kk = 0; kk < K2 / 32; ++kk) {
            const int kb = kk * 32;
            // B fragment: column n = gn*16+l16 of B == row n of Ws; K runs [kb+lhi*8, +8) and +16
            const _Float16* wp = &Ws[gn * 16 + l16][kb + lhi * 8];
            const v8h b0 = *(const v8h*)(wp);
            const v8h b1 = *(const v8h*)(wp + 16);
            const v16h bf = __builtin_shufflevector(b0, b1,
                0,1,2,3,4,5,6,7,8,9,10,11,12,13,14,15);
            // A fragments: row m = bm*16+l16; K runs [kb+lhi*8, +8) and +16
            const _Float16* ap0 = &As[q][(bm0    ) * 16 + l16][kb + lhi * 8];
            const _Float16* ap1 = &As[q][(bm0 + 1) * 16 + l16][kb + lhi * 8];
            const v8h a00 = *(const v8h*)(ap0);
            const v8h a01 = *(const v8h*)(ap0 + 16);
            const v8h a10 = *(const v8h*)(ap1);
            const v8h a11 = *(const v8h*)(ap1 + 16);
            const v16h af0 = __builtin_shufflevector(a00, a01,
                0,1,2,3,4,5,6,7,8,9,10,11,12,13,14,15);
            const v16h af1 = __builtin_shufflevector(a10, a11,
                0,1,2,3,4,5,6,7,8,9,10,11,12,13,14,15);
            acc0 = __builtin_amdgcn_wmma_f32_16x16x32_f16(
                false, af0, false, bf, (short)0, acc0, false, false);
            acc1 = __builtin_amdgcn_wmma_f32_16x16x32_f16(
                false, af1, false, bf, (short)0, acc1, false, false);
        }

        // ---- convert + store x_{t+1} into the alternate A buffer ----
        if (t + 1 < Tlen) {
            #pragma unroll
            for (int u = 0; u < 16; ++u) {
                const int i  = tid + (u << 8);
                const int bb = i >> 6;
                const int kb = (i & 63) << 2;
                v4h hv;
                hv[0] = (_Float16)xr[u].x; hv[1] = (_Float16)xr[u].y;
                hv[2] = (_Float16)xr[u].z; hv[3] = (_Float16)xr[u].w;
                *(v4h*)&As[q ^ 1][bb][kb] = hv;
            }
        }

        // ---- spill gate tiles to LDS (C/D layout: M = r + 8*lhi, N = l16) ----
        #pragma unroll
        for (int r = 0; r < 8; ++r) {
            Gl[(bm0    ) * 16 + r + 8 * lhi][gn * 16 + l16] = acc0[r];
            Gl[(bm0 + 1) * 16 + r + 8 * lhi][gn * 16 + l16] = acc1[r];
        }
        __syncthreads();

        // ---- elementwise LSTM cell update for this WG's 16 hidden units ----
        _Float16* hdst = hbuf + (q ^ 1) * (Bsz * Hdim);
        for (int i = tid; i < Bsz * HW; i += 256) {
            const int bb = i >> 4;
            const int j  = i & 15;
            const float gi = Gl[bb][ 0 + j];
            const float gf = Gl[bb][16 + j];
            const float gg = Gl[bb][32 + j];
            const float go = Gl[bb][48 + j];
            const float c  = sigmoidf_fast(gf) * Cl[bb][j]
                           + sigmoidf_fast(gi) * tanhf_fast(gg);
            Cl[bb][j] = c;
            const float h = sigmoidf_fast(go) * tanhf_fast(c);
            hdst[bb * Hdim + wg * HW + j] = (_Float16)h;
            if (t == Tlen - 1) {
                out[bb * Hdim + wg * HW + j] = h;                  // final h
                out[Bsz * Hdim + bb * Hdim + wg * HW + j] = c;     // final c
            }
        }

        // ---- grid-wide step barrier (monotonic counter in L2) ----
        __threadfence();                 // release this WG's h stores
        __syncthreads();
        if (tid == 0) {
            __hip_atomic_fetch_add(bar, 1u, __ATOMIC_RELEASE,
                                   __HIP_MEMORY_SCOPE_AGENT);
            const unsigned target = (unsigned)(t + 1) * NWG;
            while (__hip_atomic_load(bar, __ATOMIC_ACQUIRE,
                                     __HIP_MEMORY_SCOPE_AGENT) < target) {
                __builtin_amdgcn_s_sleep(1);
            }
        }
        __syncthreads();
    }
}

extern "C" void kernel_launch(void* const* d_in, const int* in_sizes, int n_in,
                              void* d_out, int out_size, void* d_ws, size_t ws_size,
                              hipStream_t stream) {
    const float* xs  = (const float*)d_in[0];   // [B, T, D]
    const float* Wih = (const float*)d_in[1];   // [4H, D]
    const float* Whh = (const float*)d_in[2];   // [4H, H]
    const float* b   = (const float*)d_in[3];   // [4H]
    float* out = (float*)d_out;                 // h then c, each [B, H]

    unsigned*  bar  = (unsigned*)d_ws;
    _Float16*  hbuf = (_Float16*)((char*)d_ws + 256);  // 2 x [B, H] f16 ping-pong

    lstm_init<<<64, 256, 0, stream>>>(bar, hbuf);
    lstm_persistent<<<NWG, 256, 0, stream>>>(xs, Wih, Whh, b, out, hbuf, bar);
    (void)in_sizes; (void)n_in; (void)out_size; (void)ws_size;
}
